// SelfAttention_36155034698510
// MI455X (gfx1250) — matrix operand order repaired and checked by reading, not compiled, not run on hardware
//
#include <hip/hip_runtime.h>
#include <hip/hip_bf16.h>

typedef __attribute__((ext_vector_type(16))) __bf16 v16bf;
typedef __attribute__((ext_vector_type(8)))  float  v8f;
typedef __attribute__((ext_vector_type(2)))  __bf16 v2bf;

union BF16Vec {
    v16bf v;
    uint4 q[2];
    unsigned w[8];
    unsigned short h[16];
};

// f32 -> bf16 by truncation (1 VALU op)
__device__ __forceinline__ unsigned short f2bf(float f) {
    return (unsigned short)(__float_as_uint(f) >> 16);
}

// pack two f32 -> packed bf16 pair (low = a, high = b)
__device__ __forceinline__ unsigned pk2bf(float a, float b) {
#if __has_builtin(__builtin_amdgcn_cvt_pk_bf16_f32)
    v2bf r = __builtin_amdgcn_cvt_pk_bf16_f32(a, b);
    return __builtin_bit_cast(unsigned, r);
#else
    // single v_perm_b32: result bytes = {b[3], b[2], a[3], a[2]}
    return __builtin_amdgcn_perm(__float_as_uint(b), __float_as_uint(a), 0x07060302u);
#endif
}

#define WMMA_BF16(A, Bm, Cm) \
    __builtin_amdgcn_wmma_f32_16x16x32_bf16(false, (A), false, (Bm), (short)0, (Cm), false, false)

// ---------------------------------------------------------------------------
// Projection: out[d,n] = sum_c w[d,c] * x[b,c,n] + bias[d], bf16 output.
// transposed==1 -> store outT[b][n][d] (row length 64, for q/k)
// transposed==0 -> store out[b][c][n]  (for v)
// Block: 256 threads = 8 waves; wave computes one 16x16 tile via WMMA bf16.
// Block tile: 64 d x 32 n.  Grid: (N/32, D/64, B).
// ---------------------------------------------------------------------------
__global__ __launch_bounds__(256)
void proj_kernel(const float* __restrict__ x,      // [B][C][N] f32
                 const float* __restrict__ wmat,   // [D][C]    f32
                 const float* __restrict__ bias,   // [D]       f32
                 unsigned short* __restrict__ outp,
                 int D, int transposed)
{
    constexpr int C = 512, N = 4096;
    const int b     = blockIdx.z;
    const int lane  = threadIdx.x & 31;
    const int wave  = threadIdx.x >> 5;
    const int lo16  = lane & 15;
    const bool hi   = lane >= 16;
    const int dtile = blockIdx.y * 64 + (wave & 3) * 16;
    const int ntile = blockIdx.x * 32 + (wave >> 2) * 16;
    const int n     = ntile + lo16;
    const float* xb = x + (size_t)b * C * N;

    v8f acc = {};
    for (int c0 = 0; c0 < C; c0 += 32) {
        BF16Vec A, Bx;
        // A operand: w rows. lane<16: K = 0..7,16..23 ; lane>=16: 8..15,24..31
        const float* wrow = wmat + (size_t)(dtile + lo16) * C + c0 + (hi ? 8 : 0);
        {
            float4 a0 = *(const float4*)(wrow);
            float4 a1 = *(const float4*)(wrow + 4);
            float4 a2 = *(const float4*)(wrow + 16);
            float4 a3 = *(const float4*)(wrow + 20);
            A.w[0] = pk2bf(a0.x, a0.y); A.w[1] = pk2bf(a0.z, a0.w);
            A.w[2] = pk2bf(a1.x, a1.y); A.w[3] = pk2bf(a1.z, a1.w);
            A.w[4] = pk2bf(a2.x, a2.y); A.w[5] = pk2bf(a2.z, a2.w);
            A.w[6] = pk2bf(a3.x, a3.y); A.w[7] = pk2bf(a3.z, a3.w);
        }
        // B operand: x columns. lane<16 col n holds K=0..7,16..23 ; lane>=16: 8..15,24..31
#pragma unroll
        for (int r = 0; r < 8; ++r) {
            int c = c0 + 2 * r + (hi ? 8 : 0) + (r >= 4 ? 8 : 0);
            Bx.w[r] = pk2bf(xb[(size_t)c * N + n], xb[(size_t)(c + 1) * N + n]);
        }
        acc = WMMA_BF16(A.v, Bx.v, acc);
    }

    float vals[8];
#pragma unroll
    for (int r = 0; r < 8; ++r) {
        int d = dtile + r + (hi ? 8 : 0);
        vals[r] = acc[r] + bias[d];
    }

    if (transposed) {
        // pack 8 consecutive d into one b128 store: outT[(b*N + n)*64 + d0..d0+7]
        union { uint4 q; unsigned w[4]; } pk;
#pragma unroll
        for (int r = 0; r < 4; ++r) pk.w[r] = pk2bf(vals[2 * r], vals[2 * r + 1]);
        unsigned short* dst = outp + ((size_t)b * N + n) * 64 + dtile + (hi ? 8 : 0);
        *(uint4*)dst = pk.q;
    } else {
#pragma unroll
        for (int r = 0; r < 8; ++r) {
            int d = dtile + r + (hi ? 8 : 0);
            outp[((size_t)b * D + d) * N + n] = f2bf(vals[r]);
        }
    }
}

// ---------------------------------------------------------------------------
// Fused flash-attention:
//   s[i,j] = q_i . k_j ;  p = softmax_i(s) ;  o[c,j] = sum_i v[c,i] p[i,j]
//   out = gamma * o + x
// Grid: (N/64, B). Block 256 = 8 waves. Block j-tile = 64 columns.
// Wave -> (j-subtile of 16) x (c-chunk of 256 channels = 16 acc tiles).
// Softmax stats per j-subtile are duplicated only 2x (the two c-chunk waves).
// ---------------------------------------------------------------------------
__global__ __launch_bounds__(256)
void attn_kernel(const float* __restrict__ x,
                 const unsigned short* __restrict__ qT,  // [B][N][64] bf16
                 const unsigned short* __restrict__ kT,  // [B][N][64] bf16
                 const unsigned short* __restrict__ vb,  // [B][C][N]  bf16
                 const float* __restrict__ gamma,
                 float* __restrict__ out)
{
    constexpr int C = 512, N = 4096, CQ = 64;
    const int b    = blockIdx.y;
    const int lane = threadIdx.x & 31;
    const int wave = threadIdx.x >> 5;
    const int lo16 = lane & 15;
    const bool hi  = lane >= 16;
    const int jsub = wave & 3;          // 4 j-subtiles of 16
    const int cch  = wave >> 2;         // 2 channel chunks of 256
    const int j0   = blockIdx.x * 64 + jsub * 16;
    const int cw   = cch * 256;

    // Loop-invariant B operand: k columns j0..j0+15, K = d (0..63, two k-steps)
    const unsigned short* kcol = kT + ((size_t)b * N + j0 + lo16) * CQ;
    BF16Vec Bk0, Bk1;
    {
        const int off = hi ? 8 : 0;
        Bk0.q[0] = *(const uint4*)(kcol + off);
        Bk0.q[1] = *(const uint4*)(kcol + off + 16);
        Bk1.q[0] = *(const uint4*)(kcol + 32 + off);
        Bk1.q[1] = *(const uint4*)(kcol + 32 + off + 16);
    }

    v8f acc[16] = {};
    float m = -INFINITY;
    float l = 0.0f;

    for (int i0 = 0; i0 < N; i0 += 32) {
        // ---- S tiles: rows i0..i0+15 (S0) and i0+16..i0+31 (S1), cols j ----
        v8f S0 = {}, S1 = {};
        {
            BF16Vec A;
            const unsigned short* qrow0 =
                qT + ((size_t)b * N + i0 + lo16) * CQ + (hi ? 8 : 0);
            A.q[0] = *(const uint4*)(qrow0);
            A.q[1] = *(const uint4*)(qrow0 + 16);
            S0 = WMMA_BF16(A.v, Bk0.v, S0);
            A.q[0] = *(const uint4*)(qrow0 + 32);
            A.q[1] = *(const uint4*)(qrow0 + 48);
            S0 = WMMA_BF16(A.v, Bk1.v, S0);

            const unsigned short* qrow1 = qrow0 + 16 * CQ;
            A.q[0] = *(const uint4*)(qrow1);
            A.q[1] = *(const uint4*)(qrow1 + 16);
            S1 = WMMA_BF16(A.v, Bk0.v, S1);
            A.q[0] = *(const uint4*)(qrow1 + 32);
            A.q[1] = *(const uint4*)(qrow1 + 48);
            S1 = WMMA_BF16(A.v, Bk1.v, S1);
            // prefetch next i-tile of qT
            __builtin_prefetch(qrow1 + 16 * CQ, 0, 0);
        }

        // ---- online softmax over i (column-wise; lane pair L / L+16 share col)
        float tmax = S0[0];
#pragma unroll
        for (int r = 0; r < 8; ++r) { tmax = fmaxf(tmax, S0[r]); tmax = fmaxf(tmax, S1[r]); }
        tmax = fmaxf(tmax, __shfl_xor(tmax, 16, 32));
        const float mnew = fmaxf(m, tmax);

        // rescale only when some column's max actually moved (rare after warmup)
        if (__any(tmax > m)) {
            float scale = __expf(m - mnew);
            l *= scale;
#pragma unroll
            for (int t = 0; t < 16; ++t)
#pragma unroll
                for (int r = 0; r < 8; ++r) acc[t][r] *= scale;
        }
        m = mnew;

        float p0[8], p1[8], tsum = 0.0f;
#pragma unroll
        for (int r = 0; r < 8; ++r) {
            p0[r] = __expf(S0[r] - mnew);
            p1[r] = __expf(S1[r] - mnew);
            tsum += p0[r] + p1[r];
        }
        tsum += __shfl_xor(tsum, 16, 32);
        l += tsum;

        // ---- repack P (32 x 16) into B-operand layout ----
        // lane<16  needs P[i0+0..15, j]   : rows 0..7 local (S0), 8..15 from partner
        // lane>=16 needs P[i0+16..31, j]  : rows 16..23 from partner (S1), 24..31 local
        // Pack to bf16 pairs FIRST (8x v_perm), then shuffle packed words (8x).
        BF16Vec P;
        {
            unsigned pw[8];
#pragma unroll
            for (int qq = 0; qq < 4; ++qq) {
                pw[qq]     = pk2bf(p0[2 * qq], p0[2 * qq + 1]);
                pw[4 + qq] = pk2bf(p1[2 * qq], p1[2 * qq + 1]);
            }
#pragma unroll
            for (int qq = 0; qq < 4; ++qq) {
                unsigned s0 = (unsigned)__shfl_xor((int)pw[qq], 16, 32);
                unsigned s1 = (unsigned)__shfl_xor((int)pw[4 + qq], 16, 32);
                P.w[qq]     = hi ? s1 : pw[qq];
                P.w[4 + qq] = hi ? pw[4 + qq] : s0;
            }
        }

        // ---- O += V * P : 16 channel sub-tiles of 16 ----
        const unsigned short* vbase =
            vb + ((size_t)b * C + cw + lo16) * N + i0 + (hi ? 16 : 0);
        __builtin_prefetch(vbase + 32, 0, 0);
#pragma unroll
        for (int t = 0; t < 16; ++t) {
            const unsigned short* vrow = vbase + (size_t)(t * 16) * N;
            BF16Vec Va;
            Va.q[0] = *(const uint4*)(vrow);
            Va.q[1] = *(const uint4*)(vrow + 8);
            acc[t] = WMMA_BF16(Va.v, P.v, acc[t]);
        }
    }

    // ---- epilogue: normalize, gamma, residual ----
    const float g  = gamma[0];
    const float rl = 1.0f / l;
#pragma unroll
    for (int t = 0; t < 16; ++t) {
#pragma unroll
        for (int r = 0; r < 8; ++r) {
            int c = cw + t * 16 + r + (hi ? 8 : 0);
            size_t idx = ((size_t)b * C + c) * N + j0 + lo16;
            out[idx] = g * (acc[t][r] * rl) + x[idx];
        }
    }
}

// ---------------------------------------------------------------------------
extern "C" void kernel_launch(void* const* d_in, const int* in_sizes, int n_in,
                              void* d_out, int out_size, void* d_ws, size_t ws_size,
                              hipStream_t stream) {
    constexpr int B = 4, C = 512, N = 4096, CQ = 64;
    const float* x     = (const float*)d_in[0];
    const float* wq    = (const float*)d_in[1];
    const float* bq    = (const float*)d_in[2];
    const float* wk    = (const float*)d_in[3];
    const float* bk    = (const float*)d_in[4];
    const float* wv    = (const float*)d_in[5];
    const float* bv    = (const float*)d_in[6];
    const float* gamma = (const float*)d_in[7];
    float* out = (float*)d_out;

    // workspace layout (bf16): qT [B][N][64], kT [B][N][64], v [B][C][N]
    unsigned short* qT = (unsigned short*)d_ws;                    //  2 MB
    unsigned short* kT = qT + (size_t)B * N * CQ;                  //  2 MB
    unsigned short* vb = kT + (size_t)B * N * CQ;                  // 16 MB

    dim3 blk(256);
    // q, k projections (D = 64, transposed store)
    proj_kernel<<<dim3(N / 32, CQ / 64, B), blk, 0, stream>>>(x, wq, bq, qT, CQ, 1);
    proj_kernel<<<dim3(N / 32, CQ / 64, B), blk, 0, stream>>>(x, wk, bk, kT, CQ, 1);
    // v projection (D = 512, row-major store)
    proj_kernel<<<dim3(N / 32, C / 64, B), blk, 0, stream>>>(x, wv, bv, vb, C, 0);
    // fused attention + residual
    attn_kernel<<<dim3(N / 64, B), blk, 0, stream>>>(x, qT, kT, vb, gamma, out);
}
